// InterpretableMultiHeadAttention_63471026700428
// MI455X (gfx1250) — matrix operand order repaired and checked by reading, not compiled, not run on hardware
//
#include <hip/hip_runtime.h>
#include <hip/hip_bf16.h>

// ---------------- problem constants ----------------
#define BATCH 16
#define TLEN  1024
#define DMOD  256
#define NH    4
#define DK    64
#define HDIM  256           // NH*DK
#define MTOT  (BATCH*TLEN)  // 16384

typedef __attribute__((ext_vector_type(16))) __bf16        v16bf;
typedef __attribute__((ext_vector_type(2)))  __bf16        v2bf;
typedef __attribute__((ext_vector_type(8)))  float         v8f;
typedef __attribute__((ext_vector_type(8)))  unsigned int  v8u;

// fp32 -> bf16 via native convert (fptrunc is RNE), returned as raw u16
static __device__ __forceinline__ unsigned short f2bf_u16(float f) {
    return __builtin_bit_cast(unsigned short, (__bf16)f);
}

// two fp32 -> packed bf16 dword (hardware v_cvt_pk_bf16_f32 when available)
static __device__ __forceinline__ unsigned cvt2_bf16(float a, float b) {
#if __has_builtin(__builtin_amdgcn_cvt_pk_bf16_f32)
    v2bf p = __builtin_amdgcn_cvt_pk_bf16_f32(a, b);
    return __builtin_bit_cast(unsigned, p);
#else
    return (unsigned)f2bf_u16(a) | ((unsigned)f2bf_u16(b) << 16);
#endif
}

// Pack two 16-byte chunks of bf16 halves into a v16bf fragment register set.
static __device__ __forceinline__ v16bf pack16(uint4 q0, uint4 q1) {
    v8u t;
    t[0] = q0.x; t[1] = q0.y; t[2] = q0.z; t[3] = q0.w;
    t[4] = q1.x; t[5] = q1.y; t[6] = q1.z; t[7] = q1.w;
    return __builtin_bit_cast(v16bf, t);
}

// A-matrix fragment (16x32 bf16): lane holds row, 16 contiguous K values
// {kb..kb+15}, kb = 16*(lane>=16). rowp points at K=0 of this row.
static __device__ __forceinline__ v16bf ld_afrag(const unsigned short* rowp, int hi) {
    const int kb = hi * 16;
    uint4 q0 = *(const uint4*)(rowp + kb);
    uint4 q1 = *(const uint4*)(rowp + kb + 8);
    return pack16(q0, q1);
}

// B-matrix fragment (32x16 bf16): lane holds column, 16 contiguous K values
// starting at 16*(lane>=16). p points at that K-run (16 halves, 16B aligned).
static __device__ __forceinline__ v16bf ld_bfrag16(const unsigned short* p) {
    uint4 q0 = *(const uint4*)(p);
    uint4 q1 = *(const uint4*)(p + 8);
    return pack16(q0, q1);
}

// CDNA5 async global->LDS DMA (16B per lane), tracked by ASYNCcnt.
// LDS address = low 32 bits of the generic pointer (LDS aperture offset).
static __device__ __forceinline__ void async_ld_b128(const void* gptr, void* lptr) {
    unsigned lds_addr = (unsigned)(size_t)lptr;
    unsigned long long gaddr = (unsigned long long)(size_t)gptr;
    asm volatile("global_load_async_to_lds_b128 %0, %1, off"
                 :: "v"(lds_addr), "v"(gaddr) : "memory");
}
static __device__ __forceinline__ void wait_async0() {
    asm volatile("s_wait_asynccnt 0x0" ::: "memory");
}

// ---------------- stage 0: weight conversion ----------------
__global__ void cvt_f32_bf16(const float* __restrict__ src,
                             unsigned short* __restrict__ dst, int n) {
    int i = (blockIdx.x * blockDim.x + threadIdx.x) * 2;
    if (i < n) *(unsigned*)(dst + i) = cvt2_bf16(src[i], src[i + 1]);
}

// ---------------- stage 1a: Q/K projections ----------------
// C = A(fp32)[M,256] @ W(bf16)[256,256] + bias, bf16 out, head-split [B,H,T,64].
// Block: 256 threads = 8 waves, tile 128(M) x 64(N), K-step 32.
__global__ __launch_bounds__(256) void proj_gemm_qk(
    const float* __restrict__ A, const unsigned short* __restrict__ W,
    const float* __restrict__ bias, unsigned short* __restrict__ Cout)
{
    __shared__ unsigned short Asm[128][40];  // [m][k], pitch 80B
    __shared__ unsigned short Bsm[64][40];   // transposed W tile: [n][k]

    const int tid  = threadIdx.x;
    const int lane = tid & 31, wave = tid >> 5;
    const int lo   = lane & 15, hi = lane >> 4;
    const int m0   = blockIdx.x * 128;
    const int n0   = blockIdx.y * 64;
    const int wm   = (wave >> 1) * 32;
    const int wn   = (wave & 1) * 32;

    v8f acc[2][2] = {};

    for (int kk = 0; kk < DMOD; kk += 32) {
        {   // A tile: 128x32 fp32 -> bf16, packed pair converts, dword LDS stores.
            const int r  = tid >> 1;
            const int c0 = (tid & 1) * 16;
            const float* src = A + (size_t)(m0 + r) * DMOD + kk + c0;
            unsigned* dst = (unsigned*)&Asm[r][c0];
#pragma unroll
            for (int i = 0; i < 8; ++i)
                dst[i] = cvt2_bf16(src[2 * i], src[2 * i + 1]);
        }
        {   // W tile: read [k][n] coalesced, store transposed [n][k].
            const int k  = tid >> 3;
            const int nb = (tid & 7) * 8;
            const unsigned short* src = W + (size_t)(kk + k) * HDIM + n0 + nb;
#pragma unroll
            for (int i = 0; i < 8; ++i) Bsm[nb + i][k] = src[i];
        }
        __syncthreads();

        v16bf af[2], bf[2];
#pragma unroll
        for (int ti = 0; ti < 2; ++ti) af[ti] = ld_afrag(&Asm[wm + ti * 16 + lo][0], hi);
#pragma unroll
        for (int tj = 0; tj < 2; ++tj) bf[tj] = ld_bfrag16(&Bsm[wn + tj * 16 + lo][hi * 16]);
#pragma unroll
        for (int ti = 0; ti < 2; ++ti)
#pragma unroll
            for (int tj = 0; tj < 2; ++tj)
                acc[ti][tj] = __builtin_amdgcn_wmma_f32_16x16x32_bf16(
                    false, af[ti], false, bf[tj], (short)0, acc[ti][tj], false, false);
        __syncthreads();
    }

#pragma unroll
    for (int ti = 0; ti < 2; ++ti)
#pragma unroll
        for (int tj = 0; tj < 2; ++tj) {
            const int col = n0 + wn + tj * 16 + lo;
            const float bv = bias[col];
            const int hh = col >> 6, d = col & (DK - 1);
#pragma unroll
            for (int r = 0; r < 8; ++r) {
                const int row  = m0 + wm + ti * 16 + r + 8 * hi;
                const int bidx = row >> 10, t = row & (TLEN - 1);
                Cout[(((size_t)bidx * NH + hh) * TLEN + t) * DK + d] =
                    f2bf_u16(acc[ti][tj][r] + bv);
            }
        }
}

// ---------------- stage 1b: V projection ----------------
// C = A(fp32)[M,256] @ W(bf16)[256,64] + bias, bf16 out, plain [M,64].
__global__ __launch_bounds__(256) void proj_gemm_v(
    const float* __restrict__ A, const unsigned short* __restrict__ W,
    const float* __restrict__ bias, unsigned short* __restrict__ Cout)
{
    __shared__ unsigned short Asm[128][40];
    __shared__ unsigned short Bsm[64][40];

    const int tid  = threadIdx.x;
    const int lane = tid & 31, wave = tid >> 5;
    const int lo   = lane & 15, hi = lane >> 4;
    const int m0   = blockIdx.x * 128;
    const int wm   = (wave >> 1) * 32;
    const int wn   = (wave & 1) * 32;

    v8f acc[2][2] = {};

    for (int kk = 0; kk < DMOD; kk += 32) {
        {
            const int r  = tid >> 1;
            const int c0 = (tid & 1) * 16;
            const float* src = A + (size_t)(m0 + r) * DMOD + kk + c0;
            unsigned* dst = (unsigned*)&Asm[r][c0];
#pragma unroll
            for (int i = 0; i < 8; ++i)
                dst[i] = cvt2_bf16(src[2 * i], src[2 * i + 1]);
        }
        {
            const int k  = tid >> 3;
            const int nb = (tid & 7) * 8;
            const unsigned short* src = W + (size_t)(kk + k) * DK + nb;
#pragma unroll
            for (int i = 0; i < 8; ++i) Bsm[nb + i][k] = src[i];
        }
        __syncthreads();

        v16bf af[2], bf[2];
#pragma unroll
        for (int ti = 0; ti < 2; ++ti) af[ti] = ld_afrag(&Asm[wm + ti * 16 + lo][0], hi);
#pragma unroll
        for (int tj = 0; tj < 2; ++tj) bf[tj] = ld_bfrag16(&Bsm[wn + tj * 16 + lo][hi * 16]);
#pragma unroll
        for (int ti = 0; ti < 2; ++ti)
#pragma unroll
            for (int tj = 0; tj < 2; ++tj)
                acc[ti][tj] = __builtin_amdgcn_wmma_f32_16x16x32_bf16(
                    false, af[ti], false, bf[tj], (short)0, acc[ti][tj], false, false);
        __syncthreads();
    }

#pragma unroll
    for (int ti = 0; ti < 2; ++ti)
#pragma unroll
        for (int tj = 0; tj < 2; ++tj) {
            const int col = wn + tj * 16 + lo;
            const float bv = bias[col];
#pragma unroll
            for (int r = 0; r < 8; ++r) {
                const int row = m0 + wm + ti * 16 + r + 8 * hi;
                Cout[(size_t)row * DK + col] = f2bf_u16(acc[ti][tj][r] + bv);
            }
        }
}

// ---------------- stage 2: flash attention ----------------
// Grid (T/128, H, B). 8 waves/block, each wave owns 16 query rows.
// K chunk staged with CDNA5 async global->LDS DMA; V transposed on the fly.
__global__ __launch_bounds__(256) void flash_attn(
    const unsigned short* __restrict__ Qp,   // [B,H,T,64] bf16
    const unsigned short* __restrict__ Kp,   // [B,H,T,64] bf16
    const unsigned short* __restrict__ Vp,   // [B,T,64]   bf16
    unsigned short* __restrict__ AO)         // [B,T,256]  bf16
{
    __shared__ unsigned short Klds[64][72];      // [tk][d]
    __shared__ unsigned short Vtlds[64][72];     // [d][tk] (transposed)
    __shared__ unsigned short Plds[8][16][72];   // per-wave P transpose scratch

    const int tid  = threadIdx.x;
    const int lane = tid & 31, wave = tid >> 5;
    const int lo   = lane & 15, hi = lane >> 4;
    const int b = blockIdx.z, h = blockIdx.y;
    const int qrow = blockIdx.x * 128 + wave * 16;
    const size_t qk_base = (size_t)(b * NH + h) * TLEN;

    // Q fragments for this wave's 16 rows (Dk=64 -> two K=32 halves), in VGPRs.
    v16bf qfrag[2];
    {
        const unsigned short* qr = Qp + (qk_base + qrow + lo) * DK;
#pragma unroll
        for (int dh = 0; dh < 2; ++dh) qfrag[dh] = ld_afrag(qr + dh * 32, hi);
    }

    float m_r[8], l_r[8];
    v8f o_acc[4] = {};
#pragma unroll
    for (int r = 0; r < 8; ++r) { m_r[r] = -1e30f; l_r[r] = 0.f; }
    const float scale = 0.125f;  // 1/sqrt(64)

    // per-thread staging coordinates
    const int srow = tid >> 2;             // 0..63
    const int scol = (tid & 3) * 16;       // 0,16,32,48

    for (int kc = 0; kc < TLEN / 64; ++kc) {
        const int tk0 = kc * 64;
        {   // K chunk -> LDS via async DMA (2 x b128 per thread)
            const unsigned short* src = Kp + (qk_base + tk0 + srow) * DK + scol;
            async_ld_b128(src,     &Klds[srow][scol]);
            async_ld_b128(src + 8, &Klds[srow][scol + 8]);
        }
        {   // V chunk -> LDS transposed [d][tk]
            const unsigned short* src = Vp + ((size_t)b * TLEN + tk0 + srow) * DK + scol;
#pragma unroll
            for (int i = 0; i < 16; ++i) Vtlds[scol + i][srow] = src[i];
        }
        if (kc + 1 < TLEN / 64) {  // prefetch next chunk into cache
            __builtin_prefetch(Kp + (qk_base + tk0 + 64 + srow) * DK + scol, 0, 0);
            __builtin_prefetch(Vp + ((size_t)b * TLEN + tk0 + 64 + srow) * DK + scol, 0, 0);
        }
        wait_async0();
        __syncthreads();

        // S = Q K^T  (16 x 64 per wave, 4 N-tiles x 2 K-halves)
        v8f s_acc[4];
#pragma unroll
        for (int n = 0; n < 4; ++n) {
            v8f s = {};
#pragma unroll
            for (int dh = 0; dh < 2; ++dh) {
                v16bf kf = ld_bfrag16(&Klds[n * 16 + lo][dh * 32 + hi * 16]);
                s = __builtin_amdgcn_wmma_f32_16x16x32_bf16(
                    false, qfrag[dh], false, kf, (short)0, s, false, false);
            }
            s_acc[n] = s;
        }
#pragma unroll
        for (int n = 0; n < 4; ++n)
#pragma unroll
            for (int r = 0; r < 8; ++r) s_acc[n][r] *= scale;

        // online softmax: row stats via xor-shuffle within 16-lane groups
        float alpha[8];
#pragma unroll
        for (int r = 0; r < 8; ++r) {
            float mx = fmaxf(fmaxf(s_acc[0][r], s_acc[1][r]),
                             fmaxf(s_acc[2][r], s_acc[3][r]));
#pragma unroll
            for (int off = 1; off < 16; off <<= 1)
                mx = fmaxf(mx, __shfl_xor(mx, off, 32));
            const float mnew = fmaxf(m_r[r], mx);
            alpha[r] = __expf(m_r[r] - mnew);
            m_r[r] = mnew;
#pragma unroll
            for (int n = 0; n < 4; ++n) o_acc[n][r] *= alpha[r];
        }

        // P = exp(S - m), row sums, and C-layout -> A-layout transpose via LDS
        float rs[8];
#pragma unroll
        for (int r = 0; r < 8; ++r) rs[r] = 0.f;
#pragma unroll
        for (int n = 0; n < 4; ++n)
#pragma unroll
            for (int r = 0; r < 8; ++r) {
                const float p = __expf(s_acc[n][r] - m_r[r]);
                rs[r] += p;
                Plds[wave][r + 8 * hi][n * 16 + lo] = f2bf_u16(p);
            }
#pragma unroll
        for (int r = 0; r < 8; ++r) {
            float s = rs[r];
#pragma unroll
            for (int off = 1; off < 16; off <<= 1) s += __shfl_xor(s, off, 32);
            l_r[r] = l_r[r] * alpha[r] + s;
        }

        // O += P @ V   (contraction over 64 keys = two K=32 halves)
#pragma unroll
        for (int kh = 0; kh < 2; ++kh) {
            v16bf pf = ld_afrag(&Plds[wave][lo][kh * 32], hi);
#pragma unroll
            for (int n = 0; n < 4; ++n) {
                v16bf vf = ld_bfrag16(&Vtlds[n * 16 + lo][kh * 32 + hi * 16]);
                o_acc[n] = __builtin_amdgcn_wmma_f32_16x16x32_bf16(
                    false, pf, false, vf, (short)0, o_acc[n], false, false);
            }
        }
        __syncthreads();
    }

    // normalize and store concat-head layout [B,T,H*64]
#pragma unroll
    for (int n = 0; n < 4; ++n)
#pragma unroll
        for (int r = 0; r < 8; ++r) {
            const int row = qrow + r + 8 * hi;
            const int d   = n * 16 + lo;
            const float v = o_acc[n][r] / l_r[r];
            AO[((size_t)b * TLEN + row) * HDIM + h * DK + d] = f2bf_u16(v);
        }
}

// ---------------- stage 3: output projection ----------------
// out(fp32)[M,256] = AO(bf16)[M,256] @ Wo(bf16)[256,256] + bo
// bf16 A tile staged with async global->LDS DMA.
__global__ __launch_bounds__(256) void out_gemm(
    const unsigned short* __restrict__ A, const unsigned short* __restrict__ W,
    const float* __restrict__ bias, float* __restrict__ Cout)
{
    __shared__ unsigned short Asm[128][40];
    __shared__ unsigned short Bsm[64][40];

    const int tid  = threadIdx.x;
    const int lane = tid & 31, wave = tid >> 5;
    const int lo   = lane & 15, hi = lane >> 4;
    const int m0   = blockIdx.x * 128;
    const int n0   = blockIdx.y * 64;
    const int wm   = (wave >> 1) * 32;
    const int wn   = (wave & 1) * 32;

    v8f acc[2][2] = {};

    for (int kk = 0; kk < HDIM; kk += 32) {
        {   // A tile (bf16): async DMA, 2 x b128 per thread
            const int r  = tid >> 1;
            const int c0 = (tid & 1) * 16;
            const unsigned short* src = A + (size_t)(m0 + r) * HDIM + kk + c0;
            async_ld_b128(src,     &Asm[r][c0]);
            async_ld_b128(src + 8, &Asm[r][c0 + 8]);
        }
        {
            const int k  = tid >> 3;
            const int nb = (tid & 7) * 8;
            const unsigned short* src = W + (size_t)(kk + k) * HDIM + n0 + nb;
#pragma unroll
            for (int i = 0; i < 8; ++i) Bsm[nb + i][k] = src[i];
        }
        wait_async0();
        __syncthreads();

        v16bf af[2], bf[2];
#pragma unroll
        for (int ti = 0; ti < 2; ++ti) af[ti] = ld_afrag(&Asm[wm + ti * 16 + lo][0], hi);
#pragma unroll
        for (int tj = 0; tj < 2; ++tj) bf[tj] = ld_bfrag16(&Bsm[wn + tj * 16 + lo][hi * 16]);
#pragma unroll
        for (int ti = 0; ti < 2; ++ti)
#pragma unroll
            for (int tj = 0; tj < 2; ++tj)
                acc[ti][tj] = __builtin_amdgcn_wmma_f32_16x16x32_bf16(
                    false, af[ti], false, bf[tj], (short)0, acc[ti][tj], false, false);
        __syncthreads();
    }

#pragma unroll
    for (int ti = 0; ti < 2; ++ti)
#pragma unroll
        for (int tj = 0; tj < 2; ++tj) {
            const int col = n0 + wn + tj * 16 + lo;
            const float bv = bias[col];
#pragma unroll
            for (int r = 0; r < 8; ++r) {
                const int row = m0 + wm + ti * 16 + r + 8 * hi;
                Cout[(size_t)row * HDIM + col] = acc[ti][tj][r] + bv;
            }
        }
}

// ---------------- launch ----------------
extern "C" void kernel_launch(void* const* d_in, const int* in_sizes, int n_in,
                              void* d_out, int out_size, void* d_ws, size_t ws_size,
                              hipStream_t stream) {
    (void)in_sizes; (void)n_in; (void)out_size; (void)ws_size;
    const float* query = (const float*)d_in[0];
    const float* key   = (const float*)d_in[1];
    const float* value = (const float*)d_in[2];
    const float* Wq = (const float*)d_in[3];
    const float* bq = (const float*)d_in[4];
    const float* Wk = (const float*)d_in[5];
    const float* bk = (const float*)d_in[6];
    const float* Wv = (const float*)d_in[7];
    const float* bv = (const float*)d_in[8];
    const float* Wo = (const float*)d_in[9];
    const float* bo = (const float*)d_in[10];
    float* out = (float*)d_out;

    // workspace layout (bf16 halves)
    unsigned short* ws    = (unsigned short*)d_ws;
    unsigned short* Wq_bf = ws;                        // 256*256
    unsigned short* Wk_bf = Wq_bf + 256 * 256;
    unsigned short* Wv_bf = Wk_bf + 256 * 256;         // 256*64
    unsigned short* Wo_bf = Wv_bf + 256 * 64;          // 256*256
    unsigned short* Qp    = Wo_bf + 256 * 256;         // B*H*T*DK
    unsigned short* Kp    = Qp + (size_t)BATCH * NH * TLEN * DK;
    unsigned short* Vp    = Kp + (size_t)BATCH * NH * TLEN * DK;  // B*T*DK
    unsigned short* AO    = Vp + (size_t)BATCH * TLEN * DK;       // B*T*HDIM

    cvt_f32_bf16<<<(256 * 256 / 2 + 255) / 256, 256, 0, stream>>>(Wq, Wq_bf, 256 * 256);
    cvt_f32_bf16<<<(256 * 256 / 2 + 255) / 256, 256, 0, stream>>>(Wk, Wk_bf, 256 * 256);
    cvt_f32_bf16<<<(256 * 64 / 2 + 255) / 256, 256, 0, stream>>>(Wv, Wv_bf, 256 * 64);
    cvt_f32_bf16<<<(256 * 256 / 2 + 255) / 256, 256, 0, stream>>>(Wo, Wo_bf, 256 * 256);

    dim3 blk(256);
    proj_gemm_qk<<<dim3(MTOT / 128, HDIM / 64), blk, 0, stream>>>(query, Wq_bf, bq, Qp);
    proj_gemm_qk<<<dim3(MTOT / 128, HDIM / 64), blk, 0, stream>>>(key,   Wk_bf, bk, Kp);
    proj_gemm_v <<<dim3(MTOT / 128, 1),         blk, 0, stream>>>(value, Wv_bf, bv, Vp);

    flash_attn<<<dim3(TLEN / 128, NH, BATCH), blk, 0, stream>>>(Qp, Kp, Vp, AO);

    out_gemm<<<dim3(MTOT / 128, HDIM / 64), blk, 0, stream>>>(AO, Wo_bf, bo, out);
}
